// tra_pre_36369783062843
// MI455X (gfx1250) — matrix operand order repaired and checked by reading, not compiled, not run on hardware
//
#include <hip/hip_runtime.h>
#include <hip/hip_bf16.h>
#include <math.h>

// Problem constants from the reference
#define BATCH 4
#define TSTEPS 32
#define NPTS 1024
#define GRAPHS (BATCH * TSTEPS)   // 128
#define HD 64
#define HS 128
#define THR2 0.09f                 // 0.3^2

typedef __attribute__((ext_vector_type(16))) _Float16 v16h;
typedef __attribute__((ext_vector_type(4)))  _Float16 v4h;
typedef __attribute__((ext_vector_type(8)))  float    v8f;

// ---------------------------------------------------------------------------
// Kernel 1: per-node degree -> dinv = 1/sqrt(deg)  (self-loop included since
// dist(i,i)=0 < thr). 4 blocks of 256 threads per graph.
// ---------------------------------------------------------------------------
__global__ __launch_bounds__(256) void k_deg(const float* __restrict__ pts,
                                             float* __restrict__ dinv) {
  int g   = blockIdx.x >> 2;
  int sub = blockIdx.x & 3;
  __shared__ float2 sp[NPTS];
  const float2* p = reinterpret_cast<const float2*>(pts) + (size_t)g * NPTS;
  for (int i = threadIdx.x; i < NPTS; i += 256) sp[i] = p[i];
  __syncthreads();
  int i = sub * 256 + threadIdx.x;
  float2 pi = sp[i];
  int cnt = 0;
  for (int j = 0; j < NPTS; ++j) {
    float dx = pi.x - sp[j].x, dy = pi.y - sp[j].y;
    cnt += (dx * dx + dy * dy < THR2) ? 1 : 0;
  }
  dinv[(size_t)g * NPTS + i] = rsqrtf((float)cnt);
}

// ---------------------------------------------------------------------------
// Kernel 2: h1 = An @ pts ; x1 = relu(h1 @ W1^T + b1) ; yT[n][j] = dinv_j*x1[j][n]
// stored FP16, transposed (feature-major) for contiguous WMMA B-operand loads.
// ---------------------------------------------------------------------------
__global__ __launch_bounds__(256) void k_feat(const float* __restrict__ pts,
                                              const float* __restrict__ dinv,
                                              const float* __restrict__ W1,
                                              const float* __restrict__ b1,
                                              _Float16* __restrict__ yT) {
  int g   = blockIdx.x >> 2;
  int sub = blockIdx.x & 3;
  __shared__ float2 sp[NPTS];
  __shared__ float  sd[NPTS];
  const float2* p  = reinterpret_cast<const float2*>(pts) + (size_t)g * NPTS;
  const float*  dv = dinv + (size_t)g * NPTS;
  for (int i = threadIdx.x; i < NPTS; i += 256) { sp[i] = p[i]; sd[i] = dv[i]; }
  __syncthreads();
  int j = sub * 256 + threadIdx.x;
  float2 pj = sp[j];
  float hx = 0.f, hy = 0.f;
  for (int i = 0; i < NPTS; ++i) {
    float dx = pj.x - sp[i].x, dy = pj.y - sp[i].y;
    if (dx * dx + dy * dy < THR2) {
      float w = sd[i];
      hx = fmaf(w, sp[i].x, hx);
      hy = fmaf(w, sp[i].y, hy);
    }
  }
  float dj = sd[j];
  hx *= dj; hy *= dj;
  _Float16* out = yT + (size_t)g * HD * NPTS + j;
  #pragma unroll 8
  for (int n = 0; n < HD; ++n) {
    float v = fmaf(W1[2 * n], hx, fmaf(W1[2 * n + 1], hy, b1[n]));
    v = fmaxf(v, 0.f) * dj;                    // fold dinv_j into y
    out[(size_t)n * NPTS] = (_Float16)v;       // coalesced across lanes
  }
}

// ---------------------------------------------------------------------------
// Kernel 3 (WMMA): h2 = An @ x1 via binary-A FP16 WMMA with f32 accumulate.
// A tile (16x32 binary adjacency) is built ONCE per block, cooperatively, in
// the exact per-lane WMMA A-operand layout in LDS (double-buffered), then all
// 4 waves fetch it with one 32B LDS read each. B operand software-pipelined.
// Epilogue: x2 = relu(h2 @ W2^T + b2), deterministic column partials.
// ---------------------------------------------------------------------------
__global__ __launch_bounds__(128) void k_gcn2(const float* __restrict__ pts,
                                              const float* __restrict__ dinv,
                                              const _Float16* __restrict__ yT,
                                              const float* __restrict__ W2,
                                              const float* __restrict__ b2,
                                              float* __restrict__ pcol) {
  int g  = blockIdx.x >> 6;
  int rt = blockIdx.x & 63;
  int m0 = rt * 16;
  __shared__ float2   sp[NPTS];          // 8 KB points
  __shared__ float    sW2[64 * 65];      // padded W2
  __shared__ float    sb2[64];
  __shared__ float    sh2[16 * 68];      // padded h2 tile
  __shared__ float    sred[128];
  __shared__ _Float16 sa[2][512];        // A operand, WMMA layout, dbl-buffered

  const float2* p = reinterpret_cast<const float2*>(pts) + (size_t)g * NPTS;
  for (int i = threadIdx.x; i < NPTS; i += 128) sp[i] = p[i];
  for (int i = threadIdx.x; i < 64 * 64; i += 128) sW2[(i >> 6) * 65 + (i & 63)] = W2[i];
  if (threadIdx.x < 64) sb2[threadIdx.x] = b2[threadIdx.x];

  int tid  = threadIdx.x;
  int lane = tid & 31;
  int wv   = tid >> 5;                   // 0..3: feature-column slab
  int m    = lane & 15;
  int hh   = lane >> 4;

  // --- cooperative A-builder mapping: thread handles 4 consecutive elements
  // of one operand lane.  Entry (alane, e) holds A[row = alane&15][K] with
  // K = (alane>>4)*16 + e  (contiguous k order, matches B).
  int alane = tid >> 2;                  // 0..31
  int ae0   = (tid & 3) * 4;             // 0,4,8,12 (e>>3 constant in quad)
  int aKb   = ((alane >> 4) << 4) + ae0;
  __syncthreads();                       // sp ready before pam read
  float2 pam = sp[m0 + (alane & 15)];

  const _Float16* yb = yT + (size_t)(g * HD + wv * 16 + m) * NPTS + hh * 16;

  auto buildA = [&](int k0, int buf) {
    // 4 consecutive points, 16B-aligned (aKb % 4 == 0)
    const float4* q = reinterpret_cast<const float4*>(&sp[k0 + aKb]);
    float4 q01 = q[0];                   // points j, j+1
    float4 q23 = q[1];                   // points j+2, j+3
    float d0x = pam.x - q01.x, d0y = pam.y - q01.y;
    float d1x = pam.x - q01.z, d1y = pam.y - q01.w;
    float d2x = pam.x - q23.x, d2y = pam.y - q23.y;
    float d3x = pam.x - q23.z, d3y = pam.y - q23.w;
    v4h hv;
    hv[0] = (_Float16)((fmaf(d0x, d0x, d0y * d0y) < THR2) ? 1.0f : 0.0f);
    hv[1] = (_Float16)((fmaf(d1x, d1x, d1y * d1y) < THR2) ? 1.0f : 0.0f);
    hv[2] = (_Float16)((fmaf(d2x, d2x, d2y * d2y) < THR2) ? 1.0f : 0.0f);
    hv[3] = (_Float16)((fmaf(d3x, d3x, d3y * d3y) < THR2) ? 1.0f : 0.0f);
    *reinterpret_cast<v4h*>(&sa[buf][alane * 16 + ae0]) = hv;   // ds_store_b64
  };

  buildA(0, 0);
  v16h b = *reinterpret_cast<const v16h*>(yb);    // preload B for k0 = 0
  __syncthreads();

  v8f acc = {};
  for (int k0 = 0; k0 < NPTS; k0 += 32) {
    int buf = (k0 >> 5) & 1;
    v16h a = *reinterpret_cast<const v16h*>(&sa[buf][lane * 16]);  // 32B LDS
    v16h bcur = b;
    if (k0 + 32 < NPTS) {                // uniform branch: EXEC stays all-1s
      b = *reinterpret_cast<const v16h*>(yb + k0 + 32);  // pipeline next B
      buildA(k0 + 32, buf ^ 1);          // fill other buffer
    }
    acc = __builtin_amdgcn_wmma_f32_16x16x32_f16(
        /*neg_a=*/false, a, /*neg_b=*/false, bcur,
        /*c_mod=*/(short)0, acc, /*reuse_a=*/false, /*reuse_b=*/false);
    __syncthreads();                     // separates read(buf) from write(buf)
  }

  // D layout: VGPR r, lane -> M = r + 8*hh, N = lane&15. Apply row dinv.
  #pragma unroll
  for (int r = 0; r < 8; ++r) {
    int row = r + 8 * hh;
    float dr = dinv[(size_t)g * NPTS + m0 + row];
    sh2[row * 68 + wv * 16 + m] = dr * acc[r];
  }
  __syncthreads();

  // x2 = relu(W2 @ h2_row + b2); partial column sums over the 16 rows.
  int c  = tid & 63;
  int rr = tid >> 6;                     // 0/1: which 8 rows
  float partial = 0.f;
  for (int r8 = 0; r8 < 8; ++r8) {
    int row = rr * 8 + r8;
    float s = sb2[c];
    for (int k = 0; k < 64; ++k) s = fmaf(sW2[c * 65 + k], sh2[row * 68 + k], s);
    partial += fmaxf(s, 0.f);
  }
  sred[tid] = partial;
  __syncthreads();
  if (tid < 64)
    pcol[(size_t)blockIdx.x * 64 + tid] = sred[tid] + sred[tid + 64];
}

// ---------------------------------------------------------------------------
// Kernel 4: reduce row-tile partials -> graph feature, Wg projection, GRU,
// final linear + time-mean. One block per batch, 128 threads (= hidden units).
// ---------------------------------------------------------------------------
__global__ __launch_bounds__(128) void k_head(const float* __restrict__ pcol,
                                              const float* __restrict__ Wg,
                                              const float* __restrict__ bg,
                                              const float* __restrict__ Wih,
                                              const float* __restrict__ Whh,
                                              const float* __restrict__ bih,
                                              const float* __restrict__ bhh,
                                              const float* __restrict__ Wf,
                                              const float* __restrict__ bf,
                                              float* __restrict__ out) {
  int b = blockIdx.x;
  __shared__ float sgs[TSTEPS][64];
  __shared__ float sx[TSTEPS][64];
  __shared__ float sh[HS];
  __shared__ float red[HS];
  int tid = threadIdx.x;

  for (int i = tid; i < TSTEPS * 64; i += 128) {
    int t = i >> 6, k = i & 63;
    const float* src = pcol + ((size_t)(b * TSTEPS + t) * 64) * 64 + k;
    float s = 0.f;
    for (int rtile = 0; rtile < 64; ++rtile) s += src[(size_t)rtile * 64];
    sgs[t][k] = s * (1.f / (float)NPTS);   // mean over N
  }
  __syncthreads();

  for (int i = tid; i < TSTEPS * 64; i += 128) {
    int t = i >> 6, n = i & 63;
    float s = bg[n];
    for (int k = 0; k < 64; ++k) s = fmaf(Wg[n * 64 + k], sgs[t][k], s);
    sx[t][n] = s;
  }
  sh[tid] = 0.f;
  __syncthreads();

  int j = tid;
  float hsum = 0.f;
  for (int t = 0; t < TSTEPS; ++t) {
    const float* x = sx[t];
    float ir = bih[j], iz = bih[HS + j], inn = bih[2 * HS + j];
    for (int k = 0; k < HD; ++k) {
      float xv = x[k];
      ir  = fmaf(Wih[(size_t)j * HD + k],            xv, ir);
      iz  = fmaf(Wih[(size_t)(HS + j) * HD + k],     xv, iz);
      inn = fmaf(Wih[(size_t)(2 * HS + j) * HD + k], xv, inn);
    }
    float hr = bhh[j], hz = bhh[HS + j], hn = bhh[2 * HS + j];
    for (int k = 0; k < HS; ++k) {
      float hv = sh[k];
      hr = fmaf(Whh[(size_t)j * HS + k],            hv, hr);
      hz = fmaf(Whh[(size_t)(HS + j) * HS + k],     hv, hz);
      hn = fmaf(Whh[(size_t)(2 * HS + j) * HS + k], hv, hn);
    }
    float r = 1.f / (1.f + expf(-(ir + hr)));
    float z = 1.f / (1.f + expf(-(iz + hz)));
    float n = tanhf(inn + r * hn);
    float hnew = (1.f - z) * n + z * sh[j];
    __syncthreads();
    sh[j] = hnew;
    hsum += hnew;
    __syncthreads();
  }

  for (int c = 0; c < 2; ++c) {
    red[tid] = Wf[(size_t)c * HS + tid] * hsum;
    __syncthreads();
    for (int s = 64; s > 0; s >>= 1) {
      if (tid < s) red[tid] += red[tid + s];
      __syncthreads();
    }
    if (tid == 0) out[b * 2 + c] = bf[c] + red[0] * (1.f / (float)TSTEPS);
    __syncthreads();
  }
}

// ---------------------------------------------------------------------------
extern "C" void kernel_launch(void* const* d_in, const int* in_sizes, int n_in,
                              void* d_out, int out_size, void* d_ws, size_t ws_size,
                              hipStream_t stream) {
  const float* pts = (const float*)d_in[0];
  const float* W1  = (const float*)d_in[1];
  const float* b1  = (const float*)d_in[2];
  const float* W2  = (const float*)d_in[3];
  const float* b2  = (const float*)d_in[4];
  const float* Wg  = (const float*)d_in[5];
  const float* bg  = (const float*)d_in[6];
  const float* Wih = (const float*)d_in[7];
  const float* Whh = (const float*)d_in[8];
  const float* bih = (const float*)d_in[9];
  const float* bhh = (const float*)d_in[10];
  const float* Wf  = (const float*)d_in[11];
  const float* bf  = (const float*)d_in[12];

  char* ws = (char*)d_ws;
  const size_t off_dinv = 0;                                        // 512 KB
  const size_t off_yT   = off_dinv + (size_t)GRAPHS * NPTS * 4;     // 16 MB
  const size_t off_pcol = off_yT + (size_t)GRAPHS * HD * NPTS * 2;  // 2 MB
  float*    dinv = (float*)(ws + off_dinv);
  _Float16* yT   = (_Float16*)(ws + off_yT);
  float*    pcol = (float*)(ws + off_pcol);

  k_deg <<<GRAPHS * 4, 256, 0, stream>>>(pts, dinv);
  k_feat<<<GRAPHS * 4, 256, 0, stream>>>(pts, dinv, W1, b1, yT);
  k_gcn2<<<GRAPHS * 64, 128, 0, stream>>>(pts, dinv, yT, W2, b2, pcol);
  k_head<<<BATCH, 128, 0, stream>>>(pcol, Wg, bg, Wih, Whh, bih, bhh, Wf, bf,
                                    (float*)d_out);
}
